// WaveNet_VNNs_36197984371251
// MI455X (gfx1250) — matrix operand order repaired and checked by reading, not compiled, not run on hardware
//
#include <hip/hip_runtime.h>
#include <hip/hip_bf16.h>
#include <math.h>

#define B_   8
#define T_   32000
#define RESC 32

typedef __attribute__((ext_vector_type(16))) _Float16 v16h;
typedef __attribute__((ext_vector_type(8)))  _Float16 v8h;
typedef __attribute__((ext_vector_type(8)))  float    v8f;
typedef __attribute__((ext_vector_type(4)))  float    v4f;

__device__ __forceinline__ float sigmoidf_(float x) { return 1.0f / (1.0f + __expf(-x)); }

// ---------------------------------------------------------------------------
// Front conv: x [B,1,T] -> h [B,T,32] (causal k=3), also zero skip buffer.
// ---------------------------------------------------------------------------
__global__ __launch_bounds__(256) void conv1_kernel(
    const float* __restrict__ x, const float* __restrict__ w,
    float* __restrict__ h, float* __restrict__ skip)
{
    int idx = blockIdx.x * 256 + threadIdx.x;   // (b,t) flat, B*T total
    int b = idx / T_;
    int t = idx - b * T_;
    const float* xb = x + (size_t)b * T_;
    float x0 = (t >= 2) ? xb[t - 2] : 0.f;
    float x1 = (t >= 1) ? xb[t - 1] : 0.f;
    float x2 = xb[t];
    float* hp = h + (size_t)idx * RESC;
    float* sp = skip + (size_t)idx * RESC;
#pragma unroll
    for (int o = 0; o < RESC; o++) {
        hp[o] = w[o * 3 + 0] * x0 + w[o * 3 + 1] * x1 + w[o * 3 + 2] * x2;
        sp[o] = 0.f;
    }
}

// ---------------------------------------------------------------------------
// One gated residual layer, WMMA f32<=f16 16x16x32 (K=32=RES channels).
//   y = dilated_conv_k3(h_in, w1)   (one K=32 WMMA per tap per 16-out tile)
//   g = tanh(y[0:32]) * sigmoid(y[32:64])
//   z = 1x1_conv(g, w2)             (WMMA)
//   h_out = h_in + z[0:32];  skip += z[32:64]
// Block = 256 threads = 8 waves; each wave owns a 16-time-step column tile.
// Weights are staged once per block into LDS, pre-swizzled into the exact
// WMMA A-fragment lane layout, so waves fetch fragments with ds_load_b128.
// ---------------------------------------------------------------------------
__global__ __launch_bounds__(256) void res_layer_kernel(
    const float* __restrict__ h_in,   // [B, T, 32]
    float* __restrict__ h_out,        // [B, T, 32]
    float* __restrict__ skip,         // [B, T, 32] running sum
    const float* __restrict__ w1,     // [64][32][3]
    const float* __restrict__ w2,     // [64][32]
    int dil)
{
    // A-fragment pool: frag f = k*4+m (dilated conv taps), 12+m (1x1 conv).
    // Per fragment: 32 lanes x 16 f16 elements = 1 KB.   Total 16 KB.
    __shared__ __align__(64) _Float16 lds_w[16][32][16];
    // Gate-output bounce buffer (D-layout f32 tile -> B-layout f16 frag). 8 KB.
    __shared__ __align__(64) _Float16 lds_g[8][16][32];

    const int tid  = threadIdx.x;
    const int wave = tid >> 5;
    const int lane = tid & 31;
    const int n    = lane & 15;     // time column within the wave's tile
    const int hi   = lane >> 4;     // which K-half this lane holds

    // ---- Stage weights: each thread fills 2 fragment-lane slots -------------
#pragma unroll
    for (int s = 0; s < 2; s++) {
        const int slot = s * 256 + tid;      // 0..511 = frag*32 + lane
        const int f  = slot >> 5;
        const int l  = slot & 31;
        const int ln = l & 15;               // M row of the fragment
        const int lh = l >> 4;               // K-half
        v16h tv = {};
        if (f < 12) {
            const int kk = f >> 2;           // tap
            const int o  = (f & 3) * 16 + ln;
#pragma unroll
            for (int e = 0; e < 16; e++) {
                int kc = lh * 16 + e;
                tv[e] = (_Float16)w1[(o * 32 + kc) * 3 + kk];
            }
        } else {
            const int o = (f - 12) * 16 + ln;
#pragma unroll
            for (int e = 0; e < 16; e++) {
                int kc = lh * 16 + e;
                tv[e] = (_Float16)w2[o * 32 + kc];
            }
        }
        *(v16h*)&lds_w[f][l][0] = tv;
    }
    __syncthreads();

    const int tpb = T_ / 128;       // time tiles per batch (block covers 128 t)
    const int b   = blockIdx.x / tpb;
    const int t   = (blockIdx.x - b * tpb) * 128 + wave * 16 + n;

    // ---- Dilated conv: 3 taps, each a K=32 WMMA against shifted columns -----
    v8f acc[4] = {v8f{}, v8f{}, v8f{}, v8f{}};
#pragma unroll
    for (int k = 0; k < 3; k++) {
        const int tc = t - (2 - k) * dil;            // causal tap position
        v16h bf = {};
        if (tc >= 0) {
            const v4f* src = (const v4f*)(h_in + ((size_t)b * T_ + tc) * RESC + hi * 16);
            v4f p0 = src[0], p1 = src[1], p2 = src[2], p3 = src[3];
#pragma unroll
            for (int e = 0; e < 4; e++) {
                bf[e]      = (_Float16)p0[e];
                bf[4 + e]  = (_Float16)p1[e];
                bf[8 + e]  = (_Float16)p2[e];
                bf[12 + e] = (_Float16)p3[e];
            }
        }
#pragma unroll
        for (int m = 0; m < 4; m++) {
            v16h a = *(const v16h*)&lds_w[k * 4 + m][lane][0];
            acc[m] = __builtin_amdgcn_wmma_f32_16x16x32_f16(false, a, false, bf,
                                                            (short)0, acc[m], false, false);
        }
    }

    // prefetch next block's tile while we do the gate math
    __builtin_prefetch(h_in + ((size_t)b * T_ + t) * RESC + 128 * RESC, 0, 1);

    // ---- Gate: g = tanh(y[0:32]) * sigmoid(y[32:64]) ------------------------
    // D layout: element r -> channel m*16 + hi*8 + r at column n.
    v8h g0, g1;
#pragma unroll
    for (int r = 0; r < 8; r++) {
        g0[r] = (_Float16)(tanhf(acc[0][r]) * sigmoidf_(acc[2][r]));   // ch hi*8+r
        g1[r] = (_Float16)(tanhf(acc[1][r]) * sigmoidf_(acc[3][r]));   // ch 16+hi*8+r
    }
    *(v8h*)&lds_g[wave][n][hi * 8]      = g0;
    *(v8h*)&lds_g[wave][n][16 + hi * 8] = g1;
    __syncthreads();   // re-stripe D-layout -> B-layout through LDS

    // B fragment for the 1x1 GEMM: lane -> N = n, K half = hi*16 + e
    v16h bg = *(const v16h*)&lds_g[wave][n][hi * 16];

    v8f z[4] = {v8f{}, v8f{}, v8f{}, v8f{}};
#pragma unroll
    for (int m = 0; m < 4; m++) {
        v16h a = *(const v16h*)&lds_w[12 + m][lane][0];
        z[m] = __builtin_amdgcn_wmma_f32_16x16x32_f16(false, a, false, bg,
                                                      (short)0, z[m], false, false);
    }

    // ---- Residual + skip stores (each lane owns 8 contiguous channels) ------
    const size_t base = ((size_t)b * T_ + t) * RESC;
    float* ho = h_out + base;
    const float* hp = h_in + base;
    float* sp = skip + base;
#pragma unroll
    for (int r = 0; r < 8; r++) {
        ho[hi * 8 + r]       = hp[hi * 8 + r]      + z[0][r];   // res ch 0-15
        ho[16 + hi * 8 + r]  = hp[16 + hi * 8 + r] + z[1][r];   // res ch 16-31
        sp[hi * 8 + r]      += z[2][r];                          // skip ch 0-15
        sp[16 + hi * 8 + r] += z[3][r];                          // skip ch 16-31
    }
}

// ---------------------------------------------------------------------------
// Post convs (small channel counts -> scalar VALU kernels, all [B,T,C] layout)
// ---------------------------------------------------------------------------
__global__ __launch_bounds__(256) void post2_kernel(
    const float* __restrict__ skip, const float* __restrict__ w, float* __restrict__ p2)
{
    int idx = blockIdx.x * 256 + threadIdx.x;
    int b = idx / T_;
    int t = idx - b * T_;
    float acc[16];
#pragma unroll
    for (int o = 0; o < 16; o++) acc[o] = 0.f;
    for (int k = 0; k < 3; k++) {
        int tc = t - 2 + k;
        if (tc < 0) continue;
        const float* sp = skip + ((size_t)b * T_ + tc) * 32;
#pragma unroll
        for (int i = 0; i < 32; i++) {
            float v = tanhf(sp[i]);          // tanh(skip_sum) feeds post2
#pragma unroll
            for (int o = 0; o < 16; o++) acc[o] += w[o * 96 + i * 3 + k] * v;
        }
    }
    float* pp = p2 + (size_t)idx * 16;
#pragma unroll
    for (int o = 0; o < 16; o++) pp[o] = tanhf(acc[o]);
}

__global__ __launch_bounds__(256) void post3_kernel(
    const float* __restrict__ p2, const float* __restrict__ w, float* __restrict__ p3)
{
    int idx = blockIdx.x * 256 + threadIdx.x;
    int b = idx / T_;
    int t = idx - b * T_;
    float acc[8];
#pragma unroll
    for (int o = 0; o < 8; o++) acc[o] = 0.f;
    for (int k = 0; k < 3; k++) {
        int tc = t - 2 + k;
        if (tc < 0) continue;
        const float* pp = p2 + ((size_t)b * T_ + tc) * 16;
#pragma unroll
        for (int i = 0; i < 16; i++) {
            float v = pp[i];
#pragma unroll
            for (int o = 0; o < 8; o++) acc[o] += w[o * 48 + i * 3 + k] * v;
        }
    }
    float* op = p3 + (size_t)idx * 8;
#pragma unroll
    for (int o = 0; o < 8; o++) op[o] = tanhf(acc[o]);
}

__global__ __launch_bounds__(256) void post4_kernel(
    const float* __restrict__ p3, const float* __restrict__ w, float* __restrict__ p4)
{
    int idx = blockIdx.x * 256 + threadIdx.x;
    int b = idx / T_;
    int t = idx - b * T_;
    float acc = 0.f;
    for (int k = 0; k < 3; k++) {
        int tc = t - 2 + k;
        if (tc < 0) continue;
        const float* pp = p3 + ((size_t)b * T_ + tc) * 8;
#pragma unroll
        for (int i = 0; i < 8; i++) acc += w[i * 3 + k] * pp[i];
    }
    p4[idx] = tanhf(acc);
}

// Quadratic Volterra head: lin = conv_k16(o1,w1); x2 = conv_k16(o1,w2[6]);
// out = lin + sum_q x2[q]*x2[q+3]
__global__ __launch_bounds__(256) void vnn_kernel(
    const float* __restrict__ o1, const float* __restrict__ w1,
    const float* __restrict__ w2, float* __restrict__ out)
{
    int idx = blockIdx.x * 256 + threadIdx.x;
    int b = idx / T_;
    int t = idx - b * T_;
    float xv[16];
#pragma unroll
    for (int j = 0; j < 16; j++) {
        int tc = t - 15 + j;
        xv[j] = (tc >= 0) ? o1[(size_t)b * T_ + tc] : 0.f;
    }
    float lin = 0.f;
#pragma unroll
    for (int j = 0; j < 16; j++) lin += w1[j] * xv[j];
    float x2[6];
#pragma unroll
    for (int c = 0; c < 6; c++) {
        float a = 0.f;
#pragma unroll
        for (int j = 0; j < 16; j++) a += w2[c * 16 + j] * xv[j];
        x2[c] = a;
    }
    out[idx] = lin + x2[0] * x2[3] + x2[1] * x2[4] + x2[2] * x2[5];
}

// ---------------------------------------------------------------------------
extern "C" void kernel_launch(void* const* d_in, const int* in_sizes, int n_in,
                              void* d_out, int out_size, void* d_ws, size_t ws_size,
                              hipStream_t stream)
{
    const float* x       = (const float*)d_in[0];
    const float* conv1_w = (const float*)d_in[1];
    const float* res_w1  = (const float*)d_in[2];   // [10,64,32,3]
    const float* res_w2  = (const float*)d_in[3];   // [10,64,32,1]
    const float* post2_w = (const float*)d_in[4];
    const float* post3_w = (const float*)d_in[5];
    const float* post4_w = (const float*)d_in[6];
    const float* vnn1_w  = (const float*)d_in[7];
    const float* vnn2_w  = (const float*)d_in[8];
    float* out = (float*)d_out;

    const size_t NT = (size_t)B_ * T_;              // 256000
    float* h0   = (float*)d_ws;                     // [B,T,32]
    float* h1   = h0 + NT * 32;                     // [B,T,32]
    float* skip = h1 + NT * 32;                     // [B,T,32]
    float* p2   = h0;                               // reuse dead h buffers
    float* p4   = h0 + NT * 16;
    float* p3   = h1;

    conv1_kernel<<<(int)(NT / 256), 256, 0, stream>>>(x, conv1_w, h0, skip);

    float* cur = h0;
    float* nxt = h1;
    for (int s = 0; s < 2; s++) {
        for (int i = 0; i < 10; i++) {
            res_layer_kernel<<<(int)(NT / 128), 256, 0, stream>>>(
                cur, nxt, skip,
                res_w1 + (size_t)i * 64 * 32 * 3,
                res_w2 + (size_t)i * 64 * 32,
                1 << i);
            float* tmp = cur; cur = nxt; nxt = tmp;
        }
    }

    post2_kernel<<<(int)(NT / 256), 256, 0, stream>>>(skip, post2_w, p2);
    post3_kernel<<<(int)(NT / 256), 256, 0, stream>>>(p2, post3_w, p3);
    post4_kernel<<<(int)(NT / 256), 256, 0, stream>>>(p3, post4_w, p4);
    vnn_kernel  <<<(int)(NT / 256), 256, 0, stream>>>(p4, vnn1_w, vnn2_w, out);
}
